// Attention_31104153157722
// MI455X (gfx1250) — compile-verified, matrix operand order checked by reading, not run on hardware
//
#include <hip/hip_runtime.h>
#include <hip/hip_bf16.h>

typedef __attribute__((ext_vector_type(16))) _Float16 v16h;
typedef __attribute__((ext_vector_type(8)))  float    v8f;
typedef __attribute__((ext_vector_type(4)))  float    v4f;
typedef __attribute__((ext_vector_type(4)))  unsigned int u32x4;
typedef __attribute__((ext_vector_type(8)))  int      i32x8;
typedef __attribute__((ext_vector_type(4)))  int      i32x4;

#define B_   2
#define L_   2048
#define DIM_ 1024
#define H_   16
#define DH_  64
#define W_   256
#define LN10000 9.210340371976184f

__device__ __forceinline__ v8f wmma16(v16h a, v16h b, v8f c) {
    return __builtin_amdgcn_wmma_f32_16x16x32_f16(
        false, a, false, b, (short)0, c, false, false);
}

// ---------------------------------------------------------------------------
// Tensor Data Mover: build D# (groups 0/1) and issue an async tile load into
// LDS.  data_size=2B elements; dims/strides in element units; tile1=0 => 1-D.
// ---------------------------------------------------------------------------
__device__ __forceinline__ void tdm_load_tile(
        unsigned lds_addr, const void* gptr,
        unsigned tile0, unsigned tile1,
        unsigned dim0, unsigned dim1,
        unsigned long long stride0) {
    unsigned long long ga = (unsigned long long)(uintptr_t)gptr;
    u32x4 g0;
    g0[0] = 1u;                                      // count=1, user descriptor
    g0[1] = lds_addr;                                // LDS byte address
    g0[2] = (unsigned)(ga & 0xffffffffu);            // global_addr[31:0]
    g0[3] = (unsigned)((ga >> 32) & 0x01ffffffu)     // global_addr[56:32]
          | (2u << 30);                              // type = 2 ("image")
    i32x8 g1;
    g1[0] = (int)(1u << 16);                         // data_size=1 (2 bytes)
    g1[1] = (int)((dim0 & 0xffffu) << 16);           // tensor_dim0[15:0]
    g1[2] = (int)(((dim0 >> 16) & 0xffffu) | ((dim1 & 0xffffu) << 16));
    g1[3] = (int)(((dim1 >> 16) & 0xffffu) | ((tile0 & 0xffffu) << 16));
    g1[4] = (int)(tile1 & 0xffffu);                  // tile_dim1, tile_dim2=0
    g1[5] = (int)(stride0 & 0xffffffffull);          // tensor_dim0_stride[31:0]
    g1[6] = (int)((stride0 >> 32) & 0xffffull);      // stride0[47:32], stride1=0
    g1[7] = 0;
    i32x4 z4 = {0, 0, 0, 0};
#if defined(__clang_major__) && __clang_major__ >= 23
    i32x8 z8 = {0, 0, 0, 0, 0, 0, 0, 0};
    __builtin_amdgcn_tensor_load_to_lds(g0, g1, z4, z4, z8, 0);
#else
    __builtin_amdgcn_tensor_load_to_lds(g0, g1, z4, z4, 0);
#endif
}

// ---------------------------------------------------------------------------
// K1: fp32 -> f16 conversion (weights)
// ---------------------------------------------------------------------------
__global__ void cvt_f32_f16(const float* __restrict__ src,
                            _Float16* __restrict__ dst, int n) {
    int i = blockIdx.x * blockDim.x + threadIdx.x;
    if (i < n) dst[i] = (_Float16)src[i];
}

// ---------------------------------------------------------------------------
// K2: fused QKV projection.  Each wave computes a 16x64 output strip (one
// head) so one A-fragment conversion feeds 4 WMMAs.
//   z=0: Q (rope -> (b,h,l,dh)),  z=1: K (rope),  z=2: V (-> (b,h,dh,l))
// ---------------------------------------------------------------------------
__global__ void qkv_proj(const float* __restrict__ q, const float* __restrict__ k,
                         const float* __restrict__ v,
                         const _Float16* __restrict__ wq16,
                         const _Float16* __restrict__ wk16,
                         const _Float16* __restrict__ wv16,
                         _Float16* __restrict__ qrope,
                         _Float16* __restrict__ krope,
                         _Float16* __restrict__ vt) {
    const int wave = threadIdx.x >> 5;
    const int lane = threadIdx.x & 31;
    const int l16  = lane & 15;
    const int grp  = lane >> 4;
    const int khalf = grp * 16;

    const int z  = blockIdx.z;
    const float*    act = (z == 0) ? q : (z == 1) ? k : v;
    const _Float16* wgt = (z == 0) ? wq16 : (z == 1) ? wk16 : wv16;

    const int row0 = blockIdx.x * 16;                  // 16 tokens
    const int n0   = (blockIdx.y * 4 + wave) * 64;     // 64 output cols = 1 head
    const size_t arow = (size_t)(row0 + l16) * DIM_;

    v8f acc[4] = {v8f{}, v8f{}, v8f{}, v8f{}};
    for (int kk = 0; kk < DIM_; kk += 32) {
        const float* ap = act + arow + kk + khalf;
        v16h a;
#pragma unroll
        for (int i = 0; i < 16; i += 4) {
            v4f f = *(const v4f*)(ap + i);
            a[i + 0] = (_Float16)f[0];
            a[i + 1] = (_Float16)f[1];
            a[i + 2] = (_Float16)f[2];
            a[i + 3] = (_Float16)f[3];
        }
#pragma unroll
        for (int f = 0; f < 4; ++f) {
            v16h b = *(const v16h*)(wgt + (size_t)(n0 + f * 16 + l16) * DIM_
                                        + kk + khalf);
            acc[f] = wmma16(a, b, acc[f]);
        }
    }

    const int bidx = row0 / L_;
    const int h    = n0 >> 6;
#pragma unroll
    for (int f = 0; f < 4; ++f) {
        const int dh = f * 16 + l16;
#pragma unroll
        for (int g = 0; g < 8; ++g) {
            const int m   = g + (grp << 3);
            const int pos = (row0 - bidx * L_) + m;
            float val = acc[f][g];
            if (z < 2) {
                const int   de  = dh & ~1;
                const float inv = __expf(-(float)de * (LN10000 / (float)DH_));
                const float ang = (float)pos * inv;
                const float c = __cosf(ang), s = __sinf(ang);
                const float p = __shfl_xor(val, 1, 32);
                const float y = (dh & 1) ? (p * s + val * c) : (val * c - p * s);
                _Float16* dst = (z == 0) ? qrope : krope;
                dst[((size_t)(bidx * H_ + h) * L_ + pos) * DH_ + dh] = (_Float16)y;
            } else {
                vt[((size_t)(bidx * H_ + h) * DH_ + dh) * L_ + pos] = (_Float16)val;
            }
        }
    }
}

// ---------------------------------------------------------------------------
// K3: windowed causal flash attention.  2 waves/block; each wave owns a
// 16-query tile.  K/V tiles are staged into LDS by the Tensor Data Mover,
// double-buffered and synchronized with s_wait_tensorcnt.
//   K tile: contiguous 32x64 f16  (1-D TDM descriptor)
//   V tile: 64 rows (dh) x 32 keys, row stride L  (2-D TDM descriptor)
// ---------------------------------------------------------------------------
__global__ void Attention_31104153157722_kernel(
        const _Float16* __restrict__ qrope,
        const _Float16* __restrict__ krope,
        const _Float16* __restrict__ vt,
        _Float16* __restrict__ attn) {
    __shared__ alignas(128) _Float16 Kbuf[2][2][32 * 64];   // [wave][buf] 4 KB
    __shared__ alignas(128) _Float16 Vbuf[2][2][64 * 32];   // [wave][buf] 4 KB
    __shared__ alignas(128) _Float16 Plds[2][16][48];

    const int wave = threadIdx.x >> 5;
    const int lane = threadIdx.x & 31;
    const int l16  = lane & 15;
    const int grp  = lane >> 4;
    const int khalf = grp * 16;

    const int qt = blockIdx.x % (L_ / 32);
    const int bh = blockIdx.x / (L_ / 32);
    const int b  = bh / H_;
    const int h  = bh % H_;
    const int q0 = qt * 32 + wave * 16;

    const _Float16* Q = qrope + (size_t)(b * H_ + h) * L_ * DH_;
    const _Float16* K = krope + (size_t)(b * H_ + h) * L_ * DH_;
    const _Float16* V = vt    + (size_t)(b * H_ + h) * DH_ * L_;

    v16h qa0 = *(const v16h*)(Q + (size_t)(q0 + l16) * DH_ +  0 + khalf);
    v16h qa1 = *(const v16h*)(Q + (size_t)(q0 + l16) * DH_ + 32 + khalf);

    float Mrow[8], Lrow[8];
    v8f   O[4];
#pragma unroll
    for (int g = 0; g < 8; ++g) { Mrow[g] = -3.0e38f; Lrow[g] = 0.0f; }
#pragma unroll
    for (int f = 0; f < 4; ++f) O[f] = v8f{};

    int jlo = q0 - W_; if (jlo < 0) jlo = 0; jlo &= ~31;
    const int jhi     = q0 + 15;
    const int nblocks = (jhi - jlo) / 32 + 1;

    // Prologue: TDM-load block 0 into buffer 0.
    tdm_load_tile((unsigned)(size_t)&Kbuf[wave][0][0], K + (size_t)jlo * DH_,
                  2048, 0, 2048, 1, 2048ull);
    tdm_load_tile((unsigned)(size_t)&Vbuf[wave][0][0], V + jlo,
                  32, 64, 2048, 64, (unsigned long long)L_);

    for (int t = 0; t < nblocks; ++t) {
        const int j0 = jlo + t * 32;
        if (t + 1 < nblocks) {
            // All ds_loads from the buffer being recycled must be complete
            // before the TDM may overwrite it.
            asm volatile("s_wait_dscnt 0" ::: "memory");
            const int nb = (t + 1) & 1;
            tdm_load_tile((unsigned)(size_t)&Kbuf[wave][nb][0],
                          K + (size_t)(j0 + 32) * DH_, 2048, 0, 2048, 1, 2048ull);
            tdm_load_tile((unsigned)(size_t)&Vbuf[wave][nb][0],
                          V + (j0 + 32), 32, 64, 2048, 64,
                          (unsigned long long)L_);
            __builtin_amdgcn_s_wait_tensorcnt(2);   // block t landed
        } else {
            __builtin_amdgcn_s_wait_tensorcnt(0);
        }
        __builtin_amdgcn_wave_barrier();

        const _Float16* kl = &Kbuf[wave][t & 1][0];
        const _Float16* vl = &Vbuf[wave][t & 1][0];

        // ---- S = Q K^T from LDS (keys j0..j0+15, j0+16..j0+31)
        v8f S[2] = {v8f{}, v8f{}};
#pragma unroll
        for (int f = 0; f < 2; ++f) {
            const _Float16* kp = kl + (size_t)(f * 16 + l16) * DH_ + khalf;
            S[f] = wmma16(qa0, *(const v16h*)(kp +  0), S[f]);
            S[f] = wmma16(qa1, *(const v16h*)(kp + 32), S[f]);
        }

        // ---- mask + online softmax + P -> LDS
#pragma unroll
        for (int g = 0; g < 8; ++g) {
            const int i  = q0 + g + (grp << 3);
            const int jA = j0 + l16;
            const int jB = j0 + 16 + l16;
            float s0 = (jA <= i && jA >= i - W_) ? S[0][g] * 0.125f : -3.0e38f;
            float s1 = (jB <= i && jB >= i - W_) ? S[1][g] * 0.125f : -3.0e38f;

            float mx = fmaxf(s0, s1);
#pragma unroll
            for (int msk = 1; msk < 16; msk <<= 1)
                mx = fmaxf(mx, __shfl_xor(mx, msk, 32));
            const float mnew  = fmaxf(Mrow[g], mx);
            const float alpha = __expf(Mrow[g] - mnew);
            Mrow[g] = mnew;
            Lrow[g] *= alpha;
#pragma unroll
            for (int f = 0; f < 4; ++f) O[f][g] *= alpha;

            const float p0 = __expf(s0 - mnew);
            const float p1 = __expf(s1 - mnew);
            float rs = p0 + p1;
#pragma unroll
            for (int msk = 1; msk < 16; msk <<= 1)
                rs += __shfl_xor(rs, msk, 32);
            Lrow[g] += rs;

            const int m = g + (grp << 3);
            Plds[wave][m][l16]      = (_Float16)p0;
            Plds[wave][m][16 + l16] = (_Float16)p1;
        }
        __builtin_amdgcn_wave_barrier();
        asm volatile("s_wait_dscnt 0" ::: "memory");

        // ---- O += P * V (P re-read in A-fragment layout, V from LDS tile)
        v16h pa = *(const v16h*)&Plds[wave][l16][khalf];
#pragma unroll
        for (int f = 0; f < 4; ++f) {
            const _Float16* vp = vl + (size_t)(f * 16 + l16) * 32 + khalf;
            O[f] = wmma16(pa, *(const v16h*)vp, O[f]);
        }
        __builtin_amdgcn_wave_barrier();
    }

    // ---- normalize + store (b, l, h*64+dh)
#pragma unroll
    for (int g = 0; g < 8; ++g) {
        const float inv = 1.0f / Lrow[g];
        const int   m   = g + (grp << 3);
        const size_t rowbase = ((size_t)b * L_ + q0 + m) * (H_ * DH_) + h * DH_;
#pragma unroll
        for (int f = 0; f < 4; ++f)
            attn[rowbase + f * 16 + l16] = (_Float16)(O[f][g] * inv);
    }
}

// ---------------------------------------------------------------------------
// K4: output projection  out = attn @ wo^T, 16x64 strip per wave
// ---------------------------------------------------------------------------
__global__ void out_proj(const _Float16* __restrict__ attn,
                         const _Float16* __restrict__ wo16,
                         float* __restrict__ out) {
    const int wave = threadIdx.x >> 5;
    const int lane = threadIdx.x & 31;
    const int l16  = lane & 15;
    const int grp  = lane >> 4;
    const int khalf = grp * 16;

    const int row0 = blockIdx.x * 16;
    const int n0   = (blockIdx.y * 4 + wave) * 64;
    const size_t arow = (size_t)(row0 + l16) * DIM_;

    v8f acc[4] = {v8f{}, v8f{}, v8f{}, v8f{}};
    for (int kk = 0; kk < DIM_; kk += 32) {
        v16h a = *(const v16h*)(attn + arow + kk + khalf);
#pragma unroll
        for (int f = 0; f < 4; ++f) {
            v16h b = *(const v16h*)(wo16 + (size_t)(n0 + f * 16 + l16) * DIM_
                                        + kk + khalf);
            acc[f] = wmma16(a, b, acc[f]);
        }
    }
#pragma unroll
    for (int f = 0; f < 4; ++f)
#pragma unroll
        for (int g = 0; g < 8; ++g) {
            const int m = g + (grp << 3);
            out[(size_t)(row0 + m) * DIM_ + n0 + f * 16 + l16] = acc[f][g];
        }
}

// ---------------------------------------------------------------------------
extern "C" void kernel_launch(void* const* d_in, const int* in_sizes, int n_in,
                              void* d_out, int out_size, void* d_ws, size_t ws_size,
                              hipStream_t stream) {
    const float* q  = (const float*)d_in[0];
    const float* k  = (const float*)d_in[1];
    const float* v  = (const float*)d_in[2];
    const float* wq = (const float*)d_in[3];
    const float* wk = (const float*)d_in[4];
    const float* wv = (const float*)d_in[5];
    const float* wo = (const float*)d_in[6];

    const size_t MB = 1ull << 20;
    char* ws = (char*)d_ws;
    _Float16* wq16  = (_Float16*)(ws +  0 * MB);
    _Float16* wk16  = (_Float16*)(ws +  2 * MB);
    _Float16* wv16  = (_Float16*)(ws +  4 * MB);
    _Float16* wo16  = (_Float16*)(ws +  6 * MB);
    _Float16* qrope = (_Float16*)(ws +  8 * MB);   // (b,h,l,dh)
    _Float16* krope = (_Float16*)(ws + 16 * MB);   // (b,h,l,dh)
    _Float16* vt    = (_Float16*)(ws + 24 * MB);   // (b,h,dh,l)
    _Float16* attn  = (_Float16*)(ws + 32 * MB);   // (b,l,h*dh)

    const int NW = DIM_ * DIM_;
    cvt_f32_f16<<<(NW + 255) / 256, 256, 0, stream>>>(wq, wq16, NW);
    cvt_f32_f16<<<(NW + 255) / 256, 256, 0, stream>>>(wk, wk16, NW);
    cvt_f32_f16<<<(NW + 255) / 256, 256, 0, stream>>>(wv, wv16, NW);
    cvt_f32_f16<<<(NW + 255) / 256, 256, 0, stream>>>(wo, wo16, NW);

    qkv_proj<<<dim3(B_ * L_ / 16, DIM_ / 256, 3), 128, 0, stream>>>(
        q, k, v, wq16, wk16, wv16, qrope, krope, vt);

    Attention_31104153157722_kernel<<<B_ * H_ * (L_ / 32), 64, 0, stream>>>(
        qrope, krope, vt, attn);

    out_proj<<<dim3(B_ * L_ / 16, DIM_ / 256), 128, 0, stream>>>(
        attn, wo16, (float*)d_out);
}
